// GPRGNN_24481313587824
// MI455X (gfx1250) — matrix operand order repaired and checked by reading, not compiled, not run on hardware
//
#include <hip/hip_runtime.h>
#include <hip/hip_bf16.h>

typedef __attribute__((ext_vector_type(16))) __bf16 v16bf;
typedef __attribute__((ext_vector_type(8)))  __bf16 v8bf;
typedef __attribute__((ext_vector_type(4)))  __bf16 v4bf;
typedef __attribute__((ext_vector_type(8)))  float  v8f;

struct bfpair { v8bf a, b; };   // 32B, bit_cast source for a v16bf fragment

#define N_NODES 100000
#define N_EDGES 1600000
#define IN_CH   512
#define HID_CH  256
#define OUT_CH  32
#define KHOPS   10

// ---------------------------------------------------------------------------
// Pre-shuffle W1/W2 into WMMA B-fragment order, split into bf16 hi/lo pairs.
// B fragment (32x16, 16-bit): lane holds column N = lane%16,
// element idx holds K = 16*(lane/16) + idx  (contiguous order used here, same as A frags).
// Fragment storage: frag*512 + lane*16 + idx  (contiguous 32B per lane).
// ---------------------------------------------------------------------------
__global__ void gpr_shuffle_w(const float* __restrict__ W1,
                              const float* __restrict__ W2,
                              __bf16* __restrict__ w1hi, __bf16* __restrict__ w1lo,
                              __bf16* __restrict__ w2hi, __bf16* __restrict__ w2lo) {
  const int NW1 = (IN_CH / 32) * (HID_CH / 16) * 512;   // 131072
  const int NW2 = (HID_CH / 32) * (OUT_CH / 16) * 512;  // 8192
  int id = blockIdx.x * blockDim.x + threadIdx.x;
  if (id < NW1) {
    int f    = id >> 9;
    int lane = (id >> 4) & 31;
    int idx  = id & 15;
    int kt = f >> 4;          // 16 N-tiles per K-tile
    int nt = f & 15;
    int K = kt * 32 + ((lane >> 4) << 4) + idx;
    int n = nt * 16 + (lane & 15);
    float w = W1[K * HID_CH + n];
    __bf16 hi = (__bf16)w;
    w1hi[id] = hi;
    w1lo[id] = (__bf16)(w - (float)hi);
  } else if (id < NW1 + NW2) {
    int id2  = id - NW1;
    int f    = id2 >> 9;
    int lane = (id2 >> 4) & 31;
    int idx  = id2 & 15;
    int kt = f >> 1;          // 2 N-tiles per K-tile
    int nt = f & 1;
    int K = kt * 32 + ((lane >> 4) << 4) + idx;
    int n = nt * 16 + (lane & 15);
    float w = W2[K * OUT_CH + n];
    __bf16 hi = (__bf16)w;
    w2hi[id2] = hi;
    w2lo[id2] = (__bf16)(w - (float)hi);
  }
}

// ---------------------------------------------------------------------------
// Fused MLP: h = relu(x@W1 + b1)@W2 + b2 via bf16x3 split-precision WMMA.
// One workgroup (4 wave32) per 16-row tile, fp32 accumulation.
// A fragment (16x32, 16-bit): lane holds row M = lane%16 and
// elements [0..15] = K(half*16 + 0..15) (matches B fragment order):
// two contiguous 16B chunks -> 2x ds_load_b128 per operand.
// C fragment (16x16 f32): c[r] = C[M = r + 8*(lane/16), N = lane%16].
// ---------------------------------------------------------------------------
__device__ __forceinline__ v16bf load_afrag(const __bf16* row, int base, int half) {
  bfpair p;
  p.a = *reinterpret_cast<const v8bf*>(row + base + half * 16);
  p.b = *reinterpret_cast<const v8bf*>(row + base + half * 16 + 8);
  return __builtin_bit_cast(v16bf, p);
}

__global__ __launch_bounds__(128)
void gpr_mlp(const float* __restrict__ x,
             const float* __restrict__ b1, const float* __restrict__ b2,
             const __bf16* __restrict__ w1hi, const __bf16* __restrict__ w1lo,
             const __bf16* __restrict__ w2hi, const __bf16* __restrict__ w2lo,
             float* __restrict__ h) {
  __shared__ __bf16 xhi[16][520];   // 520*2B = 1040B row stride: 16B-aligned rows
  __shared__ __bf16 xlo[16][520];
  __shared__ __bf16 Hhi[16][264];   // 528B row stride: 16B-aligned rows
  __shared__ __bf16 Hlo[16][264];

  const int m0   = blockIdx.x * 16;
  const int t    = threadIdx.x;
  const int lane = t & 31;
  const int wid  = t >> 5;
  const int mrow = lane & 15;
  const int half = lane >> 4;

  // ---- stage x tile (16 x 512 f32), split hi/lo bf16 ONCE into LDS ----
  {
    const float4* xg = reinterpret_cast<const float4*>(x) + (size_t)m0 * (IN_CH / 4);
#pragma unroll
    for (int i = 0; i < 16; ++i) {
      int q   = t + 128 * i;          // 0..2047 float4 chunks
      int row = q >> 7;               // 128 float4 per row
      int c4  = q & 127;
      float4 v = xg[row * (IN_CH / 4) + c4];
      v4bf hi4, lo4;
      hi4[0] = (__bf16)v.x; lo4[0] = (__bf16)(v.x - (float)hi4[0]);
      hi4[1] = (__bf16)v.y; lo4[1] = (__bf16)(v.y - (float)hi4[1]);
      hi4[2] = (__bf16)v.z; lo4[2] = (__bf16)(v.z - (float)hi4[2]);
      hi4[3] = (__bf16)v.w; lo4[3] = (__bf16)(v.w - (float)hi4[3]);
      *reinterpret_cast<v4bf*>(&xhi[row][c4 * 4]) = hi4;   // ds_store_b64
      *reinterpret_cast<v4bf*>(&xlo[row][c4 * 4]) = lo4;
    }
  }
  __syncthreads();

  // ---- GEMM1: hidden[16,256], wave wid owns N-tiles 4*wid .. 4*wid+3 ----
  v8f acc[4] = {};
#pragma unroll 2
  for (int kt = 0; kt < IN_CH / 32; ++kt) {
    v16bf ah = load_afrag(&xhi[mrow][0], kt * 32, half);   // 2x ds_load_b128
    v16bf al = load_afrag(&xlo[mrow][0], kt * 32, half);
#pragma unroll
    for (int tt = 0; tt < 4; ++tt) {
      int f = kt * (HID_CH / 16) + (wid * 4 + tt);
      v16bf bh = *((const v16bf*)w1hi + f * 32 + lane);    // 32B/lane, contiguous
      v16bf bl = *((const v16bf*)w1lo + f * 32 + lane);
      acc[tt] = __builtin_amdgcn_wmma_f32_16x16x32_bf16(false, ah, false, bh,
                                                        (short)0, acc[tt], false, false);
      acc[tt] = __builtin_amdgcn_wmma_f32_16x16x32_bf16(false, ah, false, bl,
                                                        (short)0, acc[tt], false, false);
      acc[tt] = __builtin_amdgcn_wmma_f32_16x16x32_bf16(false, al, false, bh,
                                                        (short)0, acc[tt], false, false);
    }
  }

  // ---- bias + ReLU, split hi/lo into LDS (A operand of GEMM2) ----
#pragma unroll
  for (int tt = 0; tt < 4; ++tt) {
    int col   = (wid * 4 + tt) * 16 + (lane & 15);
    float bias = b1[col];
#pragma unroll
    for (int r = 0; r < 8; ++r) {
      int M = r + half * 8;
      float vH = acc[tt][r] + bias;
      vH = vH > 0.f ? vH : 0.f;
      __bf16 hi = (__bf16)vH;
      Hhi[M][col] = hi;
      Hlo[M][col] = (__bf16)(vH - (float)hi);
    }
  }
  __syncthreads();

  // ---- GEMM2: out[16,32] = hidden@W2 + b2, waves 0/1 take one N-tile each ----
  if (wid < 2) {
    v8f c2 = {};
#pragma unroll
    for (int kt = 0; kt < HID_CH / 32; ++kt) {
      v16bf ah = load_afrag(&Hhi[mrow][0], kt * 32, half);
      v16bf al = load_afrag(&Hlo[mrow][0], kt * 32, half);
      int f = kt * 2 + wid;
      v16bf bh = *((const v16bf*)w2hi + f * 32 + lane);
      v16bf bl = *((const v16bf*)w2lo + f * 32 + lane);
      c2 = __builtin_amdgcn_wmma_f32_16x16x32_bf16(false, ah, false, bh, (short)0, c2, false, false);
      c2 = __builtin_amdgcn_wmma_f32_16x16x32_bf16(false, ah, false, bl, (short)0, c2, false, false);
      c2 = __builtin_amdgcn_wmma_f32_16x16x32_bf16(false, al, false, bh, (short)0, c2, false, false);
    }
    int n = wid * 16 + (lane & 15);
    float bias2 = b2[n];
#pragma unroll
    for (int r = 0; r < 8; ++r) {
      int M = r + half * 8;
      h[(size_t)(m0 + M) * OUT_CH + n] = c2[r] + bias2;
    }
  }
}

// ---------------------------------------------------------------------------
// Graph propagation kernels
// ---------------------------------------------------------------------------
__global__ void gpr_zero(float* __restrict__ p, int n) {
  int i = blockIdx.x * blockDim.x + threadIdx.x;
  int stride = gridDim.x * blockDim.x;
  for (; i < n; i += stride) p[i] = 0.f;
}

__global__ void gpr_deg(const int* __restrict__ ei, unsigned* __restrict__ deg) {
  int e = blockIdx.x * blockDim.x + threadIdx.x;
  if (e < N_EDGES) atomicAdd(&deg[ei[N_EDGES + e]], 1u);
}

__global__ void gpr_dinv(unsigned* __restrict__ degv) {
  int i = blockIdx.x * blockDim.x + threadIdx.x;
  if (i < N_NODES) {
    unsigned d = degv[i];
    float f = d ? rsqrtf((float)d) : 0.f;   // deg>0 -> deg>=1 -> max(deg,1)=deg
    reinterpret_cast<float*>(degv)[i] = f;
  }
}

// per-edge normalization, hoisted out of the 10-hop loop
__global__ void gpr_norm(const int* __restrict__ ei, const float* __restrict__ dinv,
                         float* __restrict__ norm) {
  int e = blockIdx.x * blockDim.x + threadIdx.x;
  if (e < N_EDGES) norm[e] = dinv[ei[e]] * dinv[ei[N_EDGES + e]];
}

__global__ void gpr_scale(const float* __restrict__ h, float* __restrict__ out,
                          const float* __restrict__ gamma) {
  int i = blockIdx.x * blockDim.x + threadIdx.x;
  if (i < N_NODES * OUT_CH) out[i] = gamma[0] * h[i];
}

// one wave32 per edge, lane = channel: coalesced gather + coalesced f32 atomics
__global__ __launch_bounds__(256)
void gpr_hop(const int* __restrict__ ei, const float* __restrict__ norm,
             const float* __restrict__ hk, float* __restrict__ hnext) {
  int gid  = blockIdx.x * 256 + threadIdx.x;
  int e    = gid >> 5;
  int lane = gid & 31;
  if (e >= N_EDGES) return;
  int r = __builtin_amdgcn_readfirstlane(ei[e]);            // wave-uniform -> SGPR
  int c = __builtin_amdgcn_readfirstlane(ei[N_EDGES + e]);
  float w = norm[e];
  float v = w * hk[(size_t)r * OUT_CH + lane];
  unsafeAtomicAdd(&hnext[(size_t)c * OUT_CH + lane], v);    // global_atomic_add_f32
}

__global__ void gpr_axpy(float* __restrict__ out, const float* __restrict__ hn,
                         const float* __restrict__ gamma, int k) {
  int i = blockIdx.x * blockDim.x + threadIdx.x;
  if (i < N_NODES * OUT_CH) out[i] += gamma[k] * hn[i];
}

// ---------------------------------------------------------------------------
extern "C" void kernel_launch(void* const* d_in, const int* in_sizes, int n_in,
                              void* d_out, int out_size, void* d_ws, size_t ws_size,
                              hipStream_t stream) {
  const float* x     = (const float*)d_in[0];
  const int*   ei    = (const int*)d_in[1];
  const float* W1    = (const float*)d_in[2];
  const float* b1    = (const float*)d_in[3];
  const float* W2    = (const float*)d_in[4];
  const float* b2    = (const float*)d_in[5];
  const float* gamma = (const float*)d_in[6];
  float* out = (float*)d_out;

  char* ws = (char*)d_ws;
  __bf16*   w1hi = (__bf16*)(ws + 0);          // 256 KB
  __bf16*   w1lo = (__bf16*)(ws + 262144);     // 256 KB
  __bf16*   w2hi = (__bf16*)(ws + 524288);     // 16 KB
  __bf16*   w2lo = (__bf16*)(ws + 540672);     // 16 KB
  unsigned* deg  = (unsigned*)(ws + 557056);   // 100000 u32, reused as dinv f32
  float*    norm = (float*)(ws + 957184);      // 6.4 MB
  float*    h0   = (float*)(ws + 7357184);     // 12.8 MB
  float*    h1   = (float*)(ws + 20157184);    // 12.8 MB  (end ~33 MB)

  const int HN = N_NODES * OUT_CH;             // 3.2M
  const int TB = 256;

  // weights -> bf16 hi/lo fragments
  gpr_shuffle_w<<<(139264 + TB - 1) / TB, TB, 0, stream>>>(W1, W2, w1hi, w1lo, w2hi, w2lo);

  // degrees -> deg_inv_sqrt (in place) -> per-edge norm
  gpr_zero<<<(N_NODES + TB - 1) / TB, TB, 0, stream>>>((float*)deg, N_NODES);
  gpr_deg<<<(N_EDGES + TB - 1) / TB, TB, 0, stream>>>(ei, deg);
  gpr_dinv<<<(N_NODES + TB - 1) / TB, TB, 0, stream>>>(deg);
  gpr_norm<<<(N_EDGES + TB - 1) / TB, TB, 0, stream>>>(ei, (const float*)deg, norm);

  // fused MLP -> h0
  gpr_mlp<<<N_NODES / 16, 128, 0, stream>>>(x, b1, b2, w1hi, w1lo, w2hi, w2lo, h0);

  // out = gamma[0] * h
  gpr_scale<<<(HN + TB - 1) / TB, TB, 0, stream>>>(h0, out, gamma);

  // K hops of normalized propagation
  float* src = h0;
  float* dst = h1;
  for (int k = 1; k <= KHOPS; ++k) {
    gpr_zero<<<4096, TB, 0, stream>>>(dst, HN);
    gpr_hop<<<(N_EDGES * 32) / TB, TB, 0, stream>>>(ei, norm, src, dst);
    gpr_axpy<<<(HN + TB - 1) / TB, TB, 0, stream>>>(out, dst, gamma, k);
    float* tmp = src; src = dst; dst = tmp;
  }
}
